// SS2D_84645215470327
// MI455X (gfx1250) — matrix operand order repaired and checked by reading, not compiled, not run on hardware
//
#include <hip/hip_runtime.h>
#include <hip/hip_bf16.h>
#include <math.h>

// ---------------- problem constants ----------------
#define BB     8
#define HH     64
#define WW     64
#define LL     (HH * WW)          // 4096
#define MROWS  (BB * LL)          // 32768 GEMM rows everywhere
#define DMODEL 96
#define NST    16                 // D_STATE
#define DI     192                // D_INNER
#define DTR    6                  // DT_RANK
#define KD     4                  // directions
#define PN     (KD * (DTR + 2 * NST))   // 152 : concat of 4 groups of 38
#define EPSLN  1e-6f

typedef __attribute__((ext_vector_type(16))) __bf16 v16bf;
typedef __attribute__((ext_vector_type(8)))  float  v8f;
typedef __attribute__((ext_vector_type(4)))  float  f32x4;

// =====================================================================
// C[M,N] = A[M,K] @ W[N,K]^T via v_wmma_f32_16x16x32_bf16.  K and N are
// compile-time: the K-loop fully unrolls into back-to-back WMMAs and the
// store epilogue becomes base + immediate-offset b32 stores in a clause.
// One wave per 16x16 tile, 4 waves per block.  Per CDNA5 ISA 7.12.2 the
// per-lane A fragment is two contiguous 8-float runs and the B fragment
// also two contiguous 8-float runs -> b128 vector loads + v_cvt_pk_bf16_f32.
// MROWS is a multiple of 16; N edge handled constexpr (only N=152 pays).
// =====================================================================
template <int KC, int NC>
__global__ __launch_bounds__(128)
void gemm_wmma_bf16(const float* __restrict__ A, const float* __restrict__ W,
                    float* __restrict__ C) {
  constexpr int TN    = (NC + 15) >> 4;
  constexpr int TOTAL = (MROWS >> 4) * TN;
  const int tileIdx = blockIdx.x * 4 + (threadIdx.x >> 5);
  if (tileIdx >= TOTAL) return;
  const int lane = threadIdx.x & 31;
  const int hi   = lane >> 4;          // lane half
  const int lo   = lane & 15;
  const int tm   = (tileIdx / TN) << 4;
  const int tn   = (tileIdx % TN) << 4;

  const int ncol = tn + lo;
  int nrow = ncol;
  if constexpr ((NC & 15) != 0) nrow = (ncol < NC) ? ncol : NC - 1;  // clamp reads
  const float* Ap = A + (size_t)(tm + lo) * KC;                      // row always < M
  const float* Wp = W + (size_t)nrow * KC;

  v8f acc = {};
#pragma unroll
  for (int k0 = 0; k0 < KC; k0 += 32) {
    // A fragment: e 0..7 -> k = hi*8+e ; e 8..15 -> k = 16+hi*8+(e-8)
    const f32x4* a0 = (const f32x4*)(Ap + k0 + hi * 8);
    const f32x4* a1 = (const f32x4*)(Ap + k0 + 16 + hi * 8);
    // B fragment: e 0..7 -> k = hi*8+e ; e 8..15 -> k = 16+hi*8+(e-8)
    const f32x4* bp0 = (const f32x4*)(Wp + k0 + hi * 8);
    const f32x4* bp1 = (const f32x4*)(Wp + k0 + 16 + hi * 8);
    f32x4 A0 = a0[0], A1 = a0[1], A2 = a1[0], A3 = a1[1];
    f32x4 B0 = bp0[0], B1 = bp0[1], B2 = bp1[0], B3 = bp1[1];
    v16bf a, b;
#pragma unroll
    for (int e = 0; e < 4; ++e) {
      a[e]      = (__bf16)A0[e];
      a[4 + e]  = (__bf16)A1[e];
      a[8 + e]  = (__bf16)A2[e];
      a[12 + e] = (__bf16)A3[e];
      b[e]      = (__bf16)B0[e];
      b[4 + e]  = (__bf16)B1[e];
      b[8 + e]  = (__bf16)B2[e];
      b[12 + e] = (__bf16)B3[e];
    }
    acc = __builtin_amdgcn_wmma_f32_16x16x32_bf16(
        /*neg_a=*/false, a, /*neg_b=*/false, b,
        /*c_mod=*/(short)0, acc, /*reuse_a=*/false, /*reuse_b=*/false);
  }

  // D layout: VGPR r -> m = tm + r + 8*hi ; n = tn + lo.
  float* Cp = C + (size_t)(tm + (hi << 3)) * NC + ncol;
  if constexpr ((NC & 15) != 0) {
    if (ncol < NC) {
#pragma unroll
      for (int r = 0; r < 8; ++r) Cp[r * NC] = acc[r];   // immediate offsets
    }
  } else {
#pragma unroll
    for (int r = 0; r < 8; ++r) Cp[r * NC] = acc[r];     // immediate offsets
  }
}

// =====================================================================
// Depthwise 3x3 conv (pad 1) + SiLU.  xz is (B*L, 2*DI); uses cols 0..DI-1.
// Output xc (B*L, DI), channel-last so the scan gathers coalesced in d.
// =====================================================================
__global__ __launch_bounds__(256)
void conv3x3_silu(const float* __restrict__ xz, const float* __restrict__ cw,
                  const float* __restrict__ cb, float* __restrict__ xc) {
  int idx = blockIdx.x * blockDim.x + threadIdx.x;
  if (idx >= BB * LL * DI) return;
  int d = idx % DI;
  int p = (idx / DI) % LL;
  int b = idx / (DI * LL);
  int h = p / WW, w = p % WW;
  float s = cb[d];
#pragma unroll
  for (int dh = 0; dh < 3; ++dh) {
#pragma unroll
    for (int dw = 0; dw < 3; ++dw) {
      int hh = h + dh - 1, ww = w + dw - 1;
      if (hh >= 0 && hh < HH && ww >= 0 && ww < WW) {
        s += xz[((size_t)(b * LL + hh * WW + ww)) * (2 * DI) + d] *
             cw[d * 9 + dh * 3 + dw];
      }
    }
  }
  xc[idx] = s / (1.f + __expf(-s));   // SiLU
}

// =====================================================================
// Selective scan: one block per (b,k); 192 lanes = one d-channel each.
// State h[16] lives in VGPRs; 4096 sequential steps.  Direction handled
// by index bijection  l -> spatial p, ys written back at p so cross-merge
// is a plain sum.  dt = softplus(dts . dtw_d + bias_d) per step.
// =====================================================================
__global__ __launch_bounds__(DI)
void scan_kernel(const float* __restrict__ xc, const float* __restrict__ proj,
                 const float* __restrict__ dtw, const float* __restrict__ dtb,
                 const float* __restrict__ A_logs, const float* __restrict__ Ds,
                 float* __restrict__ ys) {
  const int bk = blockIdx.x;          // b*KD + k
  const int b  = bk / KD;
  const int k  = bk % KD;
  const int d  = threadIdx.x;         // 0..191
  const int gd = k * DI + d;

  float w6[DTR];
#pragma unroll
  for (int r = 0; r < DTR; ++r) w6[r] = dtw[gd * DTR + r];
  const float bias = dtb[gd];
  float An[NST];
#pragma unroll
  for (int n = 0; n < NST; ++n) An[n] = -expf(A_logs[gd * NST + n]);
  const float Dv = Ds[gd];
  float h[NST];
#pragma unroll
  for (int n = 0; n < NST; ++n) h[n] = 0.f;

  const float* projB = proj + (size_t)b * LL * PN;
  const float* xcB   = xc   + (size_t)b * LL * DI;
  float*       ysB   = ys   + (size_t)bk * LL * DI;

  for (int l = 0; l < LL; ++l) {
    int t = (k >= 2) ? (LL - 1 - l) : l;                 // reversal
    int p = (k & 1) ? ((t & (WW - 1)) * WW + (t >> 6))   // wh transpose (H=W=64)
                    : t;                                  // hw
    float u = xcB[(size_t)p * DI + d];                   // coalesced in d
    const float* pr = projB + (size_t)p * PN + k * (DTR + 2 * NST);
    float dtr = bias;
#pragma unroll
    for (int r = 0; r < DTR; ++r) dtr += pr[r] * w6[r];  // wave-uniform loads
    float dt = (dtr > 20.f) ? dtr : log1pf(__expf(dtr)); // softplus
    float du = dt * u;
    float y  = 0.f;
#pragma unroll
    for (int n = 0; n < NST; ++n) {
      float Bn = pr[DTR + n];
      float Cn = pr[DTR + NST + n];
      h[n] = h[n] * __expf(dt * An[n]) + du * Bn;
      y   += h[n] * Cn;
    }
    ysB[(size_t)p * DI + d] = y + u * Dv;
  }
}

// =====================================================================
// Cross-merge (sum 4 direction buffers at spatial p) + LayerNorm over DI
// + SiLU(z) gate.  One block (192 threads) per (b,p); LDS tree reduce.
// =====================================================================
__global__ __launch_bounds__(DI)
void merge_ln_gate(const float* __restrict__ ys, const float* __restrict__ xz,
                   const float* __restrict__ nw, const float* __restrict__ nb,
                   float* __restrict__ yg) {
  const int bp = blockIdx.x;          // b*L + p
  const int b  = bp / LL;
  const int p  = bp % LL;
  const int d  = threadIdx.x;

  float v = 0.f;
#pragma unroll
  for (int k = 0; k < KD; ++k)
    v += ys[(((size_t)(b * KD + k)) * LL + p) * DI + d];

  __shared__ float red[256];
  if (d < 64) red[DI + d] = 0.f;
  red[d] = v;
  __syncthreads();
#pragma unroll
  for (int s = 128; s > 0; s >>= 1) {
    if (d < s) red[d] += red[d + s];
    __syncthreads();
  }
  const float mean = red[0] / (float)DI;
  __syncthreads();
  float dv = v - mean;
  if (d < 64) red[DI + d] = 0.f;
  red[d] = dv * dv;
  __syncthreads();
#pragma unroll
  for (int s = 128; s > 0; s >>= 1) {
    if (d < s) red[d] += red[d + s];
    __syncthreads();
  }
  const float var = red[0] / (float)DI;

  float yn = dv * rsqrtf(var + EPSLN) * nw[d] + nb[d];
  float zv = xz[(size_t)bp * (2 * DI) + DI + d];
  float zg = zv / (1.f + __expf(-zv));                   // SiLU gate
  yg[(size_t)bp * DI + d] = yn * zg;
}

// =====================================================================
extern "C" void kernel_launch(void* const* d_in, const int* in_sizes, int n_in,
                              void* d_out, int out_size, void* d_ws, size_t ws_size,
                              hipStream_t stream) {
  (void)in_sizes; (void)n_in; (void)out_size; (void)ws_size;
  const float* x     = (const float*)d_in[0];   // (B,H,W,96)
  const float* ipw   = (const float*)d_in[1];   // (384,96)
  const float* cw    = (const float*)d_in[2];   // (192,1,3,3)
  const float* cb    = (const float*)d_in[3];   // (192,)
  const float* xpw   = (const float*)d_in[4];   // (152,192)
  const float* dtw   = (const float*)d_in[5];   // (768,6)
  const float* Alogs = (const float*)d_in[6];   // (768,16)
  const float* Ds    = (const float*)d_in[7];   // (768,)
  const float* dtb   = (const float*)d_in[8];   // (768,)
  const float* nw    = (const float*)d_in[9];   // (192,)
  const float* nb    = (const float*)d_in[10];  // (192,)
  const float* opw   = (const float*)d_in[11];  // (96,192)
  float* out = (float*)d_out;

  char* ws = (char*)d_ws;
  size_t off = 0;
  float* xz   = (float*)(ws + off); off += (size_t)MROWS * 2 * DI * sizeof(float); // 50.3MB
  float* xc   = (float*)(ws + off); off += (size_t)MROWS * DI * sizeof(float);     // 25.2MB
  float* proj = (float*)(ws + off); off += (size_t)MROWS * PN * sizeof(float);     // 19.9MB
  float* ys   = (float*)(ws + off); off += (size_t)MROWS * KD * DI * sizeof(float);// 100.7MB
  float* yg   = (float*)(ws + off); off += (size_t)MROWS * DI * sizeof(float);     // 25.2MB

  auto blocks = [](int n) { return (((MROWS >> 4) * ((n + 15) / 16)) + 3) / 4; };

  // 1) in_proj: xz = x @ in_proj_w^T   (M x 384, K=96)
  gemm_wmma_bf16<DMODEL, 2 * DI><<<blocks(2 * DI), 128, 0, stream>>>(x, ipw, xz);

  // 2) depthwise conv 3x3 + SiLU -> xc (channel-last)
  conv3x3_silu<<<(BB * LL * DI + 255) / 256, 256, 0, stream>>>(xz, cw, cb, xc);

  // 3) x_proj for all 4 direction groups at once: proj = xc @ x_proj_w^T (M x 152, K=192)
  gemm_wmma_bf16<DI, PN><<<blocks(PN), 128, 0, stream>>>(xc, xpw, proj);

  // 4) selective scan, 4 directions, state in registers
  scan_kernel<<<BB * KD, DI, 0, stream>>>(xc, proj, dtw, dtb, Alogs, Ds, ys);

  // 5) cross-merge + LayerNorm + SiLU gate
  merge_ln_gate<<<MROWS, DI, 0, stream>>>(ys, xz, nw, nb, yg);

  // 6) out_proj: out = yg @ out_proj_w^T (M x 96, K=192)
  gemm_wmma_bf16<DI, DMODEL><<<blocks(DMODEL), 128, 0, stream>>>(yg, opw, out);
}
